// ConcatAttentionHeads_24756191494476
// MI455X (gfx1250) — matrix operand order repaired and checked by reading, not compiled, not run on hardware
//
#include <hip/hip_runtime.h>
#include <hip/hip_bf16.h>

// ConcatAttentionHeads for MI455X (gfx1250): bf16 WMMA pipeline, f32 accumulate,
// double-buffered async global->LDS tile staging.
// B=4, S=2048, NHID=1024, NHEAD=16, HDIM=64.
#define B_    4
#define S_    2048
#define NHID_ 1024
#define NHEAD_ 16
#define HDIM_ 64

typedef __attribute__((ext_vector_type(8)))  __bf16 v8bf;
typedef __attribute__((ext_vector_type(16))) __bf16 v16bf;
typedef __attribute__((ext_vector_type(8)))  float  v8f;

static __device__ __forceinline__ v16bf cat16(v8bf lo, v8bf hi) {
  return __builtin_shufflevector(lo, hi, 0,1,2,3,4,5,6,7,8,9,10,11,12,13,14,15);
}

static __device__ __forceinline__ v8f wmma_bf16(v16bf a, v16bf b, v8f c) {
  // D = A(16x32 bf16) * B(32x16 bf16) + C(16x16 f32)
  return __builtin_amdgcn_wmma_f32_16x16x32_bf16(false, a, false, b, (short)0, c,
                                                 false, false);
}

static __device__ __forceinline__ __bf16 to_bf(float f) {
  // round-to-nearest-even f32 -> bf16 (bit level)
  union { float f; unsigned u; } a; a.f = f;
  unsigned r = a.u + 0x7FFFu + ((a.u >> 16) & 1u);
  unsigned short s = (unsigned short)(r >> 16);
  __bf16 out; __builtin_memcpy(&out, &s, 2);
  return out;
}

// Async DMA: 16 B per lane, global -> LDS, no VGPR round-trip (ASYNCcnt).
// LDS address = low 32 bits of the generic pointer (flat aperture mapping).
static __device__ __forceinline__ void async_ld_b128(const void* g, void* l) {
  asm volatile("global_load_async_to_lds_b128 %0, %1, off"
               :: "v"((unsigned)(unsigned long long)l), "v"(g)
               : "memory");
}

template <int N>
static __device__ __forceinline__ void wait_async() {
#if __has_builtin(__builtin_amdgcn_s_wait_asynccnt)
  __builtin_amdgcn_s_wait_asynccnt(N);
#else
  asm volatile("s_wait_asynccnt %0" :: "i"(N) : "memory");
#endif
}

// ---------------------------------------------------------------- converts
__global__ void __launch_bounds__(256) cvt_f32_bf16(const float* __restrict__ in,
                                                    __bf16* __restrict__ out, int n) {
  int i = blockIdx.x * 256 + threadIdx.x;
  if (i < n) out[i] = to_bf(in[i]);
}

// W [H][NHID][D] f32  ->  Wt [H][D][NHID] bf16  (B^T layout for WMMA B-frags)
__global__ void __launch_bounds__(256) cvt_w_tr(const float* __restrict__ in,
                                                __bf16* __restrict__ out) {
  int i   = blockIdx.x * 256 + threadIdx.x;   // H*NHID*D = 1M
  int h   = i >> 16;                          // NHID*D = 65536
  int rem = i & 65535;
  int n   = rem >> 6;
  int d   = rem & 63;
  out[(h << 16) + (d << 10) + n] = to_bf(in[i]);
}

// ---------------------------------------------------------------- QKV GEMM
// C[m, d] = sum_c xbf[m, c] * W[h, c, d] + bias[h, d], m = b*S + s
// grid = (B*S/128, NHEAD, 3); block = 256 (8 waves); tile 128x64, K-chunk 64.
// Double-buffered LDS; 6 async b128 copies per thread per tile.
__global__ void __launch_bounds__(256)
qkv_gemm(const __bf16* __restrict__ xbf,
         const __bf16* __restrict__ Wqt, const __bf16* __restrict__ Wkt,
         const __bf16* __restrict__ Wvt,
         const float* __restrict__ bq, const float* __restrict__ bk,
         const float* __restrict__ bv,
         __bf16* __restrict__ Qd, __bf16* __restrict__ Kd, __bf16* __restrict__ Vd) {
  const int tid  = threadIdx.x;
  const int wave = tid >> 5;
  const int lane = tid & 31;
  const int hi16 = lane >> 4;        // lane half (0: lanes 0-15, 1: lanes 16-31)
  const int l16  = lane & 15;
  const int m0   = blockIdx.x * 128;
  const int h    = blockIdx.y;

  const __bf16* Wt; const float* bias; __bf16* dst;
  if (blockIdx.z == 0)      { Wt = Wqt; bias = bq; dst = Qd; }
  else if (blockIdx.z == 1) { Wt = Wkt; bias = bk; dst = Kd; }
  else                      { Wt = Wvt; bias = bv; dst = Vd; }
  const __bf16* Wh = Wt + (size_t)h * 64 * NHID_;

  __shared__ __align__(16) __bf16 lA[2][128][72];  // 128 x 64 (+8 pad), x2 bufs
  __shared__ __align__(16) __bf16 lB[2][64][72];   // B^T: 64 x 64 (+8 pad), x2

  // per-thread chunk coordinates (reused every tile)
  int arow[4], acol[4], brow[2], bcol[2];
#pragma unroll
  for (int i = 0; i < 4; ++i) {
    int cid = tid + i * 256; arow[i] = cid >> 3; acol[i] = (cid & 7) * 8;
  }
#pragma unroll
  for (int i = 0; i < 2; ++i) {
    int cid = tid + i * 256; brow[i] = cid >> 3; bcol[i] = (cid & 7) * 8;
  }

  v8f acc[4] = {v8f{}, v8f{}, v8f{}, v8f{}};

  // prologue: issue tile 0
#pragma unroll
  for (int i = 0; i < 4; ++i)
    async_ld_b128(xbf + (size_t)(m0 + arow[i]) * NHID_ + acol[i], &lA[0][arow[i]][acol[i]]);
#pragma unroll
  for (int i = 0; i < 2; ++i)
    async_ld_b128(Wh + (size_t)brow[i] * NHID_ + bcol[i], &lB[0][brow[i]][bcol[i]]);

  const int NT = NHID_ / 64;  // 16 tiles
  for (int it = 0; it < NT; ++it) {
    const int cur = it & 1;
    if (it + 1 < NT) {
      const int nxt = (it + 1) & 1, kc = (it + 1) * 64;
#pragma unroll
      for (int i = 0; i < 4; ++i)
        async_ld_b128(xbf + (size_t)(m0 + arow[i]) * NHID_ + kc + acol[i],
                      &lA[nxt][arow[i]][acol[i]]);
#pragma unroll
      for (int i = 0; i < 2; ++i)
        async_ld_b128(Wh + (size_t)brow[i] * NHID_ + kc + bcol[i],
                      &lB[nxt][brow[i]][bcol[i]]);
      wait_async<6>();      // tile `it` landed; tile `it+1` still in flight
    } else {
      wait_async<0>();
    }
    __syncthreads();

    const int ar = wave * 16 + l16;
#pragma unroll
    for (int kk = 0; kk < 64; kk += 32) {
      v16bf afrag = cat16(*(const v8bf*)&lA[cur][ar][kk + hi16 * 8],
                          *(const v8bf*)&lA[cur][ar][kk + 16 + hi16 * 8]);
      v16bf bf[4];
#pragma unroll
      for (int f = 0; f < 4; ++f) {
        const int br = f * 16 + l16;
        bf[f] = cat16(*(const v8bf*)&lB[cur][br][kk + hi16 * 8],
                      *(const v8bf*)&lB[cur][br][kk + 16 + hi16 * 8]);
      }
#pragma unroll
      for (int f = 0; f < 4; ++f) acc[f] = wmma_bf16(afrag, bf[f], acc[f]);
    }
    __syncthreads();        // all waves done reading `cur` before it is reused
  }

#pragma unroll
  for (int f = 0; f < 4; ++f) {
    const int d  = f * 16 + l16;
    const float bia = bias[h * 64 + d];
#pragma unroll
    for (int r = 0; r < 8; ++r) {
      const int m = m0 + wave * 16 + r + hi16 * 8;
      const int b = m >> 11, s = m & (S_ - 1);
      dst[(((size_t)(b * NHEAD_ + h)) * S_ + s) * 64 + d] = to_bf(acc[f][r] + bia);
    }
  }
}

// ---------------------------------------------------------------- attention
// Flash-style: each wave owns a 16-row Q strip; workgroup streams 64-wide
// K/V tiles through LDS. Online softmax in f32; all GEMMs via bf16 WMMA.
// grid = (S/128, B*NHEAD); block = 256.
__global__ void __launch_bounds__(256)
attn_kernel(const __bf16* __restrict__ Q, const __bf16* __restrict__ Kk,
            const __bf16* __restrict__ V, __bf16* __restrict__ O /*[B][S][1024]*/) {
  const int tid  = threadIdx.x;
  const int wave = tid >> 5;
  const int lane = tid & 31;
  const int hi16 = lane >> 4;
  const int l16  = lane & 15;
  const int s0   = blockIdx.x * 128;
  const int bh   = blockIdx.y;
  const int b    = bh >> 4;
  const int h    = bh & 15;
  const size_t base = (size_t)bh * S_ * 64;
  const __bf16* Qg = Q  + base;
  const __bf16* Kg = Kk + base;
  const __bf16* Vg = V  + base;

  __shared__ __align__(16) __bf16 lK[64][72];       // K tile  [t][d]
  __shared__ __align__(16) __bf16 lV[64][72];       // V tile transposed [d][t]
  __shared__ __align__(16) __bf16 lP[8][16][72];    // per-wave P re-pack scratch

  // Q strip A-fragments (row = l16 within strip; K split per ISA A layout)
  const int qrow = s0 + wave * 16 + l16;
  const __bf16* qp = Qg + (size_t)qrow * 64;
  v16bf qa0 = cat16(*(const v8bf*)(qp + hi16 * 8),      *(const v8bf*)(qp + 16 + hi16 * 8));
  v16bf qa1 = cat16(*(const v8bf*)(qp + 32 + hi16 * 8), *(const v8bf*)(qp + 48 + hi16 * 8));

  float mrun[8], lrun[8];
#pragma unroll
  for (int r = 0; r < 8; ++r) { mrun[r] = -3.0e38f; lrun[r] = 0.0f; }
  v8f acc[4] = {v8f{}, v8f{}, v8f{}, v8f{}};
  const float scale = 0.125f;  // 1/sqrt(64)

  for (int t0 = 0; t0 < S_; t0 += 64) {
    __syncthreads();
    // K tile: async DMA straight to LDS. V tile: b128 load + register transpose.
    v8bf vch[2];
#pragma unroll
    for (int i = 0; i < 2; ++i) {
      int cid = tid + i * 256;
      int tt = cid >> 3, doff = (cid & 7) * 8;
      async_ld_b128(Kg + (size_t)(t0 + tt) * 64 + doff, &lK[tt][doff]);
      vch[i] = *(const v8bf*)(Vg + (size_t)(t0 + tt) * 64 + doff);
    }
#pragma unroll
    for (int i = 0; i < 2; ++i) {
      int cid = tid + i * 256;
      int tt = cid >> 3, doff = (cid & 7) * 8;
#pragma unroll
      for (int j = 0; j < 8; ++j) lV[doff + j][tt] = vch[i][j];
    }
    wait_async<0>();
    __syncthreads();

    // S = Q . K^T  (B^T rows = K rows [t][d])
    v8f sfrag[4];
#pragma unroll
    for (int f = 0; f < 4; ++f) {
      const int trow = f * 16 + l16;
      v16bf b0 = cat16(*(const v8bf*)&lK[trow][hi16 * 8],
                       *(const v8bf*)&lK[trow][16 + hi16 * 8]);
      v16bf b1 = cat16(*(const v8bf*)&lK[trow][32 + hi16 * 8],
                       *(const v8bf*)&lK[trow][48 + hi16 * 8]);
      v8f c = v8f{};
      c = wmma_bf16(qa0, b0, c);
      c = wmma_bf16(qa1, b1, c);
      sfrag[f] = c;
    }

    // online softmax: per-row max / sum via 16-lane butterflies
    float mnew[8];
#pragma unroll
    for (int f = 0; f < 4; ++f)
#pragma unroll
      for (int r = 0; r < 8; ++r) sfrag[f][r] *= scale;
#pragma unroll
    for (int r = 0; r < 8; ++r) {
      float mx = fmaxf(fmaxf(sfrag[0][r], sfrag[1][r]),
                       fmaxf(sfrag[2][r], sfrag[3][r]));
#pragma unroll
      for (int off = 1; off < 16; off <<= 1)
        mx = fmaxf(mx, __shfl_xor(mx, off, 32));
      mnew[r] = fmaxf(mrun[r], mx);
    }
#pragma unroll
    for (int f = 0; f < 4; ++f)
#pragma unroll
      for (int r = 0; r < 8; ++r)
        sfrag[f][r] = __expf(sfrag[f][r] - mnew[r]);
#pragma unroll
    for (int r = 0; r < 8; ++r) {
      float sm = sfrag[0][r] + sfrag[1][r] + sfrag[2][r] + sfrag[3][r];
#pragma unroll
      for (int off = 1; off < 16; off <<= 1)
        sm += __shfl_xor(sm, off, 32);
      const float cf = __expf(mrun[r] - mnew[r]);
      lrun[r] = lrun[r] * cf + sm;
      mrun[r] = mnew[r];
#pragma unroll
      for (int f = 0; f < 4; ++f) acc[f][r] *= cf;
    }

    // re-pack P (C layout -> A layout) through per-wave LDS scratch
#pragma unroll
    for (int f = 0; f < 4; ++f)
#pragma unroll
      for (int r = 0; r < 8; ++r)
        lP[wave][r + hi16 * 8][f * 16 + l16] = to_bf(sfrag[f][r]);
    __syncthreads();
    v16bf pa0 = cat16(*(const v8bf*)&lP[wave][l16][hi16 * 8],
                      *(const v8bf*)&lP[wave][l16][16 + hi16 * 8]);
    v16bf pa1 = cat16(*(const v8bf*)&lP[wave][l16][32 + hi16 * 8],
                      *(const v8bf*)&lP[wave][l16][48 + hi16 * 8]);

    // O += P . V   (B^T rows = V^T rows [d][t])
#pragma unroll
    for (int f = 0; f < 4; ++f) {
      const int drow = f * 16 + l16;
      v16bf b0 = cat16(*(const v8bf*)&lV[drow][hi16 * 8],
                       *(const v8bf*)&lV[drow][16 + hi16 * 8]);
      v16bf b1 = cat16(*(const v8bf*)&lV[drow][32 + hi16 * 8],
                       *(const v8bf*)&lV[drow][48 + hi16 * 8]);
      acc[f] = wmma_bf16(pa0, b0, acc[f]);
      acc[f] = wmma_bf16(pa1, b1, acc[f]);
    }
  }

  // normalize and store O as bf16 [B][S][H*64]
#pragma unroll
  for (int f = 0; f < 4; ++f) {
    const int d = f * 16 + l16;
    const int c = h * 64 + d;
#pragma unroll
    for (int r = 0; r < 8; ++r) {
      const int s = s0 + wave * 16 + r + hi16 * 8;
      O[((size_t)b * S_ + s) * (NHEAD_ * HDIM_) + c] = to_bf(acc[f][r] / lrun[r]);
    }
  }
}

// ---------------------------------------------------------------- out proj
// out[b][o][s] = sum_c Obf[b*S+s][c] * WO_w[o][c] + WO_b[o]
// WO_w row-major [o][c] is already the B^T layout. grid = (B*S/128, 1024/64).
__global__ void __launch_bounds__(256)
proj_gemm(const __bf16* __restrict__ Obf, const __bf16* __restrict__ WOt,
          const float* __restrict__ WOb, float* __restrict__ out) {
  const int tid  = threadIdx.x;
  const int wave = tid >> 5;
  const int lane = tid & 31;
  const int hi16 = lane >> 4;
  const int l16  = lane & 15;
  const int m0   = blockIdx.x * 128;
  const int n0   = blockIdx.y * 64;

  __shared__ __align__(16) __bf16 lA[2][128][72];
  __shared__ __align__(16) __bf16 lB[2][64][72];

  int arow[4], acol[4], brow[2], bcol[2];
#pragma unroll
  for (int i = 0; i < 4; ++i) {
    int cid = tid + i * 256; arow[i] = cid >> 3; acol[i] = (cid & 7) * 8;
  }
#pragma unroll
  for (int i = 0; i < 2; ++i) {
    int cid = tid + i * 256; brow[i] = cid >> 3; bcol[i] = (cid & 7) * 8;
  }

  v8f acc[4] = {v8f{}, v8f{}, v8f{}, v8f{}};

#pragma unroll
  for (int i = 0; i < 4; ++i)
    async_ld_b128(Obf + (size_t)(m0 + arow[i]) * NHID_ + acol[i], &lA[0][arow[i]][acol[i]]);
#pragma unroll
  for (int i = 0; i < 2; ++i)
    async_ld_b128(WOt + (size_t)(n0 + brow[i]) * NHID_ + bcol[i], &lB[0][brow[i]][bcol[i]]);

  const int NT = NHID_ / 64;
  for (int it = 0; it < NT; ++it) {
    const int cur = it & 1;
    if (it + 1 < NT) {
      const int nxt = (it + 1) & 1, kc = (it + 1) * 64;
#pragma unroll
      for (int i = 0; i < 4; ++i)
        async_ld_b128(Obf + (size_t)(m0 + arow[i]) * NHID_ + kc + acol[i],
                      &lA[nxt][arow[i]][acol[i]]);
#pragma unroll
      for (int i = 0; i < 2; ++i)
        async_ld_b128(WOt + (size_t)(n0 + brow[i]) * NHID_ + kc + bcol[i],
                      &lB[nxt][brow[i]][bcol[i]]);
      wait_async<6>();
    } else {
      wait_async<0>();
    }
    __syncthreads();

    const int ar = wave * 16 + l16;
#pragma unroll
    for (int kk = 0; kk < 64; kk += 32) {
      v16bf afrag = cat16(*(const v8bf*)&lA[cur][ar][kk + hi16 * 8],
                          *(const v8bf*)&lA[cur][ar][kk + 16 + hi16 * 8]);
      v16bf bf[4];
#pragma unroll
      for (int f = 0; f < 4; ++f) {
        const int br = f * 16 + l16;
        bf[f] = cat16(*(const v8bf*)&lB[cur][br][kk + hi16 * 8],
                      *(const v8bf*)&lB[cur][br][kk + 16 + hi16 * 8]);
      }
#pragma unroll
      for (int f = 0; f < 4; ++f) acc[f] = wmma_bf16(afrag, bf[f], acc[f]);
    }
    __syncthreads();
  }

#pragma unroll
  for (int f = 0; f < 4; ++f) {
    const int o = n0 + f * 16 + l16;
    const float bia = WOb[o];
#pragma unroll
    for (int r = 0; r < 8; ++r) {
      const int m = m0 + wave * 16 + r + hi16 * 8;
      const int b = m >> 11, s = m & (S_ - 1);
      out[((size_t)b * NHID_ + o) * S_ + s] = acc[f][r] + bia;
    }
  }
}

// ---------------------------------------------------------------- launcher
extern "C" void kernel_launch(void* const* d_in, const int* in_sizes, int n_in,
                              void* d_out, int out_size, void* d_ws, size_t ws_size,
                              hipStream_t stream) {
  const float* x    = (const float*)d_in[0];
  const float* Wq   = (const float*)d_in[1];
  const float* bq   = (const float*)d_in[2];
  const float* Wk   = (const float*)d_in[3];
  const float* bk   = (const float*)d_in[4];
  const float* Wv   = (const float*)d_in[5];
  const float* bv   = (const float*)d_in[6];
  const float* WO_w = (const float*)d_in[7];
  const float* WO_b = (const float*)d_in[8];

  char* ws = (char*)d_ws;
  const size_t NTOK  = (size_t)B_ * S_;             // 8192
  const size_t XBYT  = NTOK * NHID_ * 2;            // 16 MB per bf16 activation
  const size_t WBYT  = (size_t)NHEAD_ * NHID_ * HDIM_ * 2;  // 2 MB per weight

  __bf16* xbf = (__bf16*)(ws);
  __bf16* Qb  = (__bf16*)(ws + XBYT * 1);
  __bf16* Kb  = (__bf16*)(ws + XBYT * 2);
  __bf16* Vb  = (__bf16*)(ws + XBYT * 3);
  __bf16* Ob  = (__bf16*)(ws + XBYT * 4);
  __bf16* Wqt = (__bf16*)(ws + XBYT * 5);
  __bf16* Wkt = (__bf16*)(ws + XBYT * 5 + WBYT * 1);
  __bf16* Wvt = (__bf16*)(ws + XBYT * 5 + WBYT * 2);
  __bf16* WOt = (__bf16*)(ws + XBYT * 5 + WBYT * 3);

  const int nx = (int)(NTOK * NHID_);               // 8388608
  cvt_f32_bf16<<<nx / 256, 256, 0, stream>>>(x, xbf, nx);
  const int nw = NHEAD_ * NHID_ * HDIM_;            // 1048576
  cvt_w_tr<<<nw / 256, 256, 0, stream>>>(Wq, Wqt);
  cvt_w_tr<<<nw / 256, 256, 0, stream>>>(Wk, Wkt);
  cvt_w_tr<<<nw / 256, 256, 0, stream>>>(Wv, Wvt);
  cvt_f32_bf16<<<(NHID_ * NHID_) / 256, 256, 0, stream>>>(WO_w, WOt, NHID_ * NHID_);

  qkv_gemm<<<dim3((unsigned)(NTOK / 128), NHEAD_, 3), 256, 0, stream>>>(
      xbf, Wqt, Wkt, Wvt, bq, bk, bv, Qb, Kb, Vb);

  attn_kernel<<<dim3(S_ / 128, B_ * NHEAD_), 256, 0, stream>>>(Qb, Kb, Vb, Ob);

  proj_gemm<<<dim3((unsigned)(NTOK / 128), NHID_ / 64), 256, 0, stream>>>(
      Ob, WOt, WO_b, (float*)d_out);
}
